// Attention_28458453303813
// MI455X (gfx1250) — matrix operand order repaired and checked by reading, not compiled, not run on hardware
//
#include <hip/hip_runtime.h>
#include <stdint.h>

// Problem constants (match reference)
#define BATCH   8
#define SEQ     2048
#define DIM     512
#define NHEADS  8
#define HDIM    64
#define QK_SCALE 0.125f   // HEAD_DIM^-0.5 = 64^-0.5

typedef __attribute__((ext_vector_type(16))) __bf16 v16bf;
typedef __attribute__((ext_vector_type(8)))  float  v8f;
typedef __attribute__((ext_vector_type(4)))  int    v4i;

union Frag {
    uint4 u[2];
    v16bf v;
};

__device__ __forceinline__ v8f wmma_bf16(v16bf a, v16bf b, v8f c) {
    // D = A(16x32 bf16) * B(32x16 bf16) + C(16x16 f32)
    return __builtin_amdgcn_wmma_f32_16x16x32_bf16(
        /*neg_a=*/false, a, /*neg_b=*/false, b,
        /*c_mod=*/(short)0, c, /*reuse_a=*/false, /*reuse_b=*/false);
}

__device__ __forceinline__ unsigned short f2bf(float f) {
    unsigned int u = __float_as_uint(f);
    u += 0x7FFFu + ((u >> 16) & 1u);   // round-to-nearest-even
    return (unsigned short)(u >> 16);
}

// ---------------------------------------------------------------------------
// Async global->LDS 16-byte copy (GLOBAL_LOAD_ASYNC_TO_LDS_B128, ASYNCcnt),
// with a synchronous fallback if the builtin is absent on this toolchain.
// Builtin signature (from hipcc diagnostic): (v4i AS1*, v4i AS3*, Imm, Imm).
// ---------------------------------------------------------------------------
#if defined(__has_builtin)
#  if __has_builtin(__builtin_amdgcn_global_load_async_to_lds_b128)
#    define HAVE_ASYNC_LDS 1
#  endif
#endif

typedef __attribute__((address_space(1))) v4i* gv4i_ptr;
typedef __attribute__((address_space(3))) v4i* lv4i_ptr;

__device__ __forceinline__ void copy16B_g2lds(unsigned short* dst_lds,
                                              const unsigned short* src_glb) {
#ifdef HAVE_ASYNC_LDS
    __builtin_amdgcn_global_load_async_to_lds_b128(
        (gv4i_ptr)src_glb, (lv4i_ptr)dst_lds, /*imm offset=*/0, /*cpol=*/0);
#else
    *(uint4*)dst_lds = *(const uint4*)src_glb;
#endif
}

__device__ __forceinline__ void wait_async_then_barrier() {
#ifdef HAVE_ASYNC_LDS
#  if __has_builtin(__builtin_amdgcn_s_wait_asynccnt)
    __builtin_amdgcn_s_wait_asynccnt(0);
#  else
    asm volatile("s_wait_asynccnt 0" ::: "memory");
#  endif
#endif
    __syncthreads();
}

// ---------------------------------------------------------------------------
// Kernel 1: QKV projection.  out[m,e] = sum_c x[m,c] * qkv_w[e,c]
// M = B*N = 16384, K = 512, E = 1536.  Block computes a 128(M) x 64(E) tile.
// Results scattered to bf16 q/k/v in [B,H,N,D] layout; q pre-scaled.
// ---------------------------------------------------------------------------
__global__ __launch_bounds__(256)
void qkv_gemm(const float* __restrict__ x, const float* __restrict__ w,
              unsigned short* __restrict__ qd, unsigned short* __restrict__ kd,
              unsigned short* __restrict__ vd) {
    __shared__ __align__(16) unsigned short Xs[128][40]; // 128 x 32 bf16, padded
    __shared__ __align__(16) unsigned short Ws[64][40];  //  64 x 32 bf16, padded

    const int wgM = blockIdx.x;           // 128 M-tiles
    const int wgE = blockIdx.y;           // 24 E-tiles of 64
    const int t = threadIdx.x;
    const int lane = t & 31, wv = t >> 5;
    const int wm = wv >> 2, we = wv & 3;  // wave grid 2(M) x 4(E)
    const int l15 = lane & 15, half = lane >> 4;

    v8f acc[4] = {};
    for (int kk = 0; kk < 512; kk += 32) {
        __syncthreads();
        { // stage X tile 128x32 (f32 -> bf16)
            int row = t >> 1, cb = (t & 1) * 16;
            const float* src = x + (size_t)(wgM * 128 + row) * 512 + kk + cb;
            unsigned short tmp[16];
#pragma unroll
            for (int j = 0; j < 16; ++j) tmp[j] = f2bf(src[j]);
            *(uint4*)&Xs[row][cb]     = *(uint4*)&tmp[0];
            *(uint4*)&Xs[row][cb + 8] = *(uint4*)&tmp[8];
        }
        { // stage W tile 64x32 (f32 -> bf16)
            int row = t >> 2, cb = (t & 3) * 8;
            const float* src = w + (size_t)(wgE * 64 + row) * 512 + kk + cb;
            unsigned short tmp[8];
#pragma unroll
            for (int j = 0; j < 8; ++j) tmp[j] = f2bf(src[j]);
            *(uint4*)&Ws[row][cb] = *(uint4*)&tmp[0];
        }
        __syncthreads();

        // B fragment (Wt): lane = output col e; 16 contiguous c per lane-half
        Frag bf;
        const unsigned short* wr = &Ws[we * 16 + l15][0];
        bf.u[0] = *(const uint4*)&wr[half * 16];
        bf.u[1] = *(const uint4*)&wr[half * 16 + 8];
#pragma unroll
        for (int i = 0; i < 4; ++i) {
            Frag af;
            const unsigned short* xr = &Xs[wm * 64 + i * 16 + l15][0];
            af.u[0] = *(const uint4*)&xr[half * 16];       // K 0-7 / 16-23
            af.u[1] = *(const uint4*)&xr[half * 16 + 8];   // K 8-15 / 24-31
            acc[i] = wmma_bf16(af.v, bf.v, acc[i]);
        }
    }

    // Epilogue: scatter to q/k/v  (E tile of 64 never straddles a 512 boundary)
    const int e_base = wgE * 64;
    const int which = e_base / 512;           // 0=q 1=k 2=v (uniform)
    const int h = (e_base % 512) / 64;        // uniform
    const int d = we * 16 + l15;
    unsigned short* dst = (which == 0) ? qd : (which == 1) ? kd : vd;
    const float scl = (which == 0) ? QK_SCALE : 1.0f;
#pragma unroll
    for (int i = 0; i < 4; ++i) {
        int m0 = wgM * 128 + wm * 64 + i * 16 + half * 8;
#pragma unroll
        for (int r = 0; r < 8; ++r) {
            int m = m0 + r;
            int b = m >> 11, n = m & (SEQ - 1);
            dst[(((b * NHEADS + h) * SEQ) + n) * HDIM + d] = f2bf(acc[i][r] * scl);
        }
    }
}

// ---------------------------------------------------------------------------
// Kernel 2: flash attention.  One block per (b*H+h, 128-query block).
// 8 waves, each owning 16 query rows; 64-key tiles; online softmax.
// ---------------------------------------------------------------------------
__global__ __launch_bounds__(256)
void flash_attn(const unsigned short* __restrict__ q,
                const unsigned short* __restrict__ k,
                const unsigned short* __restrict__ v,
                const unsigned char* __restrict__ mask,
                unsigned short* __restrict__ ctx) {
    __shared__ __align__(16) unsigned short Kt[64][72];      // [key][d]
    __shared__ __align__(16) unsigned short Vt[64][72];      // [d][key] (transposed)
    __shared__ __align__(16) unsigned short Pls[8][16][72];  // per-wave P staging

    const int bh = blockIdx.x;   // b*H + h  (64)
    const int qb = blockIdx.y;   // query block of 128 (16)
    const int b  = bh >> 3;
    const int t = threadIdx.x, lane = t & 31, wv = t >> 5;
    const int l15 = lane & 15, half = lane >> 4;

    const unsigned short* qp = q + (size_t)bh * SEQ * HDIM;
    const unsigned short* kp = k + (size_t)bh * SEQ * HDIM;
    const unsigned short* vp = v + (size_t)bh * SEQ * HDIM;
    const unsigned char*  mp = mask + b * SEQ;

    // Q fragments for this wave's 16 rows (held in registers the whole kernel)
    const int qrow = qb * 128 + wv * 16 + l15;
    const unsigned short* qr = qp + qrow * HDIM;
    Frag aq0, aq1;
    aq0.u[0] = *(const uint4*)&qr[half * 16];
    aq0.u[1] = *(const uint4*)&qr[half * 16 + 8];
    aq1.u[0] = *(const uint4*)&qr[32 + half * 16];
    aq1.u[1] = *(const uint4*)&qr[32 + half * 16 + 8];

    v8f o[4] = {};
    float mrow[8], lrow[8];
#pragma unroll
    for (int r = 0; r < 8; ++r) { mrow[r] = -1e30f; lrow[r] = 0.0f; }

    for (int kb = 0; kb < SEQ; kb += 64) {
        __syncthreads();
        { // stage K tile row-major (async DMA to LDS when available)
            int row = t >> 2, cb = (t & 3) * 16;
            const unsigned short* src = kp + (kb + row) * HDIM + cb;
            copy16B_g2lds(&Kt[row][cb],     src);
            copy16B_g2lds(&Kt[row][cb + 8], src + 8);
        }
        { // stage V tile transposed: Vt[d][key]  (needs element shuffle -> manual)
            int key = t >> 2, db = (t & 3) * 16;
            const unsigned short* src = vp + (kb + key) * HDIM + db;
            unsigned short hb[16];
            *(uint4*)&hb[0] = *(const uint4*)&src[0];
            *(uint4*)&hb[8] = *(const uint4*)&src[8];
#pragma unroll
            for (int j = 0; j < 16; ++j) Vt[db + j][key] = hb[j];
        }
        if (kb + 64 < SEQ) {   // prefetch next tiles -> global_prefetch_b8
            __builtin_prefetch(kp + (kb + 64) * HDIM + lane * 64, 0, 0);
            __builtin_prefetch(vp + (kb + 64) * HDIM + lane * 64, 0, 0);
        }
        wait_async_then_barrier();

        // S = Q * K^T  (K-dim = d = 64 -> two WMMAs per 16x16 tile)
        v8f s[4];
#pragma unroll
        for (int nt = 0; nt < 4; ++nt) {
            Frag b0, b1;
            const unsigned short* kr = &Kt[nt * 16 + l15][0];
            b0.u[0] = *(const uint4*)&kr[half * 16];
            b0.u[1] = *(const uint4*)&kr[half * 16 + 8];
            b1.u[0] = *(const uint4*)&kr[32 + half * 16];
            b1.u[1] = *(const uint4*)&kr[32 + half * 16 + 8];
            v8f z = {};
            z = wmma_bf16(aq0.v, b0.v, z);
            z = wmma_bf16(aq1.v, b1.v, z);
            // additive key-mask bias (mask is per key position)
            float bias = mp[kb + nt * 16 + l15] ? 0.0f : -1e9f;
#pragma unroll
            for (int r = 0; r < 8; ++r) z[r] += bias;
            s[nt] = z;
        }

        // Online softmax: each lane holds rows (half*8 + r), cols across 16 lanes
        float alpha[8];
#pragma unroll
        for (int r = 0; r < 8; ++r) {
            float rm = fmaxf(fmaxf(s[0][r], s[1][r]), fmaxf(s[2][r], s[3][r]));
#pragma unroll
            for (int off = 1; off < 16; off <<= 1)
                rm = fmaxf(rm, __shfl_xor(rm, off, 32));
            float mnew = fmaxf(mrow[r], rm);
            alpha[r] = __expf(mrow[r] - mnew);
            mrow[r] = mnew;
            float rs = 0.0f;
#pragma unroll
            for (int nt = 0; nt < 4; ++nt) {
                float p = __expf(s[nt][r] - mnew);
                s[nt][r] = p;
                rs += p;
            }
#pragma unroll
            for (int off = 1; off < 16; off <<= 1)
                rs += __shfl_xor(rs, off, 32);
            lrow[r] = lrow[r] * alpha[r] + rs;
        }
#pragma unroll
        for (int nt = 0; nt < 4; ++nt)
#pragma unroll
            for (int r = 0; r < 8; ++r) o[nt][r] *= alpha[r];

        // Reshape P (C layout) -> A layout via per-wave LDS round trip
#pragma unroll
        for (int nt = 0; nt < 4; ++nt)
#pragma unroll
            for (int r = 0; r < 8; ++r)
                Pls[wv][half * 8 + r][nt * 16 + l15] = f2bf(s[nt][r]);
        __syncthreads();

        // O += P * V   (K-dim = key = 64 -> two key sub-windows)
#pragma unroll
        for (int s2 = 0; s2 < 2; ++s2) {
            Frag pa;
            const unsigned short* pr = &Pls[wv][l15][s2 * 32];
            pa.u[0] = *(const uint4*)&pr[half * 16];
            pa.u[1] = *(const uint4*)&pr[half * 16 + 8];
#pragma unroll
            for (int nt = 0; nt < 4; ++nt) {
                Frag bv;
                const unsigned short* vr = &Vt[nt * 16 + l15][s2 * 32];
                bv.u[0] = *(const uint4*)&vr[half * 16];
                bv.u[1] = *(const uint4*)&vr[half * 16 + 8];
                o[nt] = wmma_bf16(pa.v, bv.v, o[nt]);
            }
        }
    }

    // Finalize: divide by l, store ctx as bf16 in [B, N, H*D] layout
    const int hh = bh & 7;
#pragma unroll
    for (int r = 0; r < 8; ++r) {
        float inv = (lrow[r] > 0.0f) ? (1.0f / lrow[r]) : 0.0f;
        int n = qb * 128 + wv * 16 + half * 8 + r;
        unsigned short* dst = ctx + ((size_t)(b * SEQ + n)) * DIM + hh * HDIM;
#pragma unroll
        for (int nt = 0; nt < 4; ++nt)
            dst[nt * 16 + l15] = f2bf(o[nt][r] * inv);
    }
}

// ---------------------------------------------------------------------------
// Kernel 3: output projection.  out[m,o] = sum_c ctx[m,c]*proj_w[o,c] + b[o]
// ---------------------------------------------------------------------------
__global__ __launch_bounds__(256)
void proj_gemm(const unsigned short* __restrict__ ctx, const float* __restrict__ w,
               const float* __restrict__ bias, float* __restrict__ out) {
    __shared__ __align__(16) unsigned short Xs[128][40];
    __shared__ __align__(16) unsigned short Ws[64][40];

    const int wgM = blockIdx.x;           // 128
    const int wgE = blockIdx.y;           // 8
    const int t = threadIdx.x;
    const int lane = t & 31, wv = t >> 5;
    const int wm = wv >> 2, we = wv & 3;
    const int l15 = lane & 15, half = lane >> 4;

    v8f acc[4] = {};
    for (int kk = 0; kk < 512; kk += 32) {
        __syncthreads();
        { // stage ctx tile (already bf16 -> async DMA straight into LDS)
            int row = t >> 1, cb = (t & 1) * 16;
            const unsigned short* src = ctx + (size_t)(wgM * 128 + row) * 512 + kk + cb;
            copy16B_g2lds(&Xs[row][cb],     src);
            copy16B_g2lds(&Xs[row][cb + 8], src + 8);
        }
        { // stage proj_w tile (f32 -> bf16)
            int row = t >> 2, cb = (t & 3) * 8;
            const float* src = w + (size_t)(wgE * 64 + row) * 512 + kk + cb;
            unsigned short tmp[8];
#pragma unroll
            for (int j = 0; j < 8; ++j) tmp[j] = f2bf(src[j]);
            *(uint4*)&Ws[row][cb] = *(uint4*)&tmp[0];
        }
        wait_async_then_barrier();

        Frag bf;
        const unsigned short* wr = &Ws[we * 16 + l15][0];
        bf.u[0] = *(const uint4*)&wr[half * 16];
        bf.u[1] = *(const uint4*)&wr[half * 16 + 8];
#pragma unroll
        for (int i = 0; i < 4; ++i) {
            Frag af;
            const unsigned short* xr = &Xs[wm * 64 + i * 16 + l15][0];
            af.u[0] = *(const uint4*)&xr[half * 16];
            af.u[1] = *(const uint4*)&xr[half * 16 + 8];
            acc[i] = wmma_bf16(af.v, bf.v, acc[i]);
        }
    }

    const int ocol = wgE * 64 + we * 16 + l15;
    const float bval = bias[ocol];
#pragma unroll
    for (int i = 0; i < 4; ++i) {
        int m0 = wgM * 128 + wm * 64 + i * 16 + half * 8;
#pragma unroll
        for (int r = 0; r < 8; ++r)
            out[(size_t)(m0 + r) * 512 + ocol] = acc[i][r] + bval;
    }
}

// ---------------------------------------------------------------------------
// Launch: qkv -> flash attention -> projection.
// Workspace: q,k,v,ctx as bf16, each B*H*N*D = 8,388,608 elements (16.8 MB);
// total 67 MB of d_ws.
// ---------------------------------------------------------------------------
extern "C" void kernel_launch(void* const* d_in, const int* in_sizes, int n_in,
                              void* d_out, int out_size, void* d_ws, size_t ws_size,
                              hipStream_t stream) {
    const float*         x      = (const float*)d_in[0];
    const unsigned char* mask   = (const unsigned char*)d_in[1]; // jax bool -> 1 byte
    const float*         qkv_w  = (const float*)d_in[2];
    const float*         proj_w = (const float*)d_in[3];
    const float*         proj_b = (const float*)d_in[4];
    float*               out    = (float*)d_out;

    const size_t SZ = (size_t)BATCH * NHEADS * SEQ * HDIM;   // == B*N*C
    unsigned short* qd  = (unsigned short*)d_ws;
    unsigned short* kd  = qd + SZ;
    unsigned short* vd  = kd + SZ;
    unsigned short* ctx = vd + SZ;

    dim3 g1(128, 24);  // M/128 x E/64
    qkv_gemm<<<g1, 256, 0, stream>>>(x, qkv_w, qd, kd, vd);

    dim3 g2(BATCH * NHEADS, SEQ / 128);
    flash_attn<<<g2, 256, 0, stream>>>(qd, kd, vd, mask, ctx);

    dim3 g3(128, 8);   // M/128 x O/64
    proj_gemm<<<g3, 256, 0, stream>>>(ctx, proj_w, proj_b, out);
}
